// QBinaryConnectNet_27161373180010
// MI455X (gfx1250) — 8-bit matrix operand order repaired and checked by reading, compiled, not run on hardware
//
#include <hip/hip_runtime.h>
#include <stdint.h>


typedef __attribute__((ext_vector_type(8))) int v8i_t;

#define EPSBN 1e-5f

__global__ void k_sign_i8(const float* __restrict__ w, int8_t* __restrict__ o, int n) {
  int i = blockIdx.x * blockDim.x + threadIdx.x;
  if (i < n) o[i] = (w[i] >= 0.f) ? (int8_t)1 : (int8_t)-1;
}

__global__ void k_conv0_bn_sign(const float* __restrict__ x, const float* __restrict__ w,
                                const float* __restrict__ g, const float* __restrict__ bb,
                                const float* __restrict__ mm, const float* __restrict__ vv,
                                int8_t* __restrict__ a0) {
  const int HO = 30, WO = 30, C = 128;
  int idx = blockIdx.x * blockDim.x + threadIdx.x;
  int total = 128 * C * HO * WO;
  if (idx >= total) return;
  int xw = idx % WO;
  int yh = (idx / WO) % HO;
  int c  = (idx / (WO * HO)) % C;
  int b  = idx / (WO * HO * C);
  float s = 0.f;
  for (int ci = 0; ci < 3; ++ci)
    for (int ky = 0; ky < 3; ++ky)
      for (int kx = 0; kx < 3; ++kx)
        s += x[((b * 3 + ci) * 32 + yh + ky) * 32 + xw + kx] *
             w[((c * 3 + ci) * 3 + ky) * 3 + kx];
  float sc = g[c] * rsqrtf(vv[c] + EPSBN);
  float f = s * sc + (bb[c] - mm[c] * sc);
  a0[idx] = (f >= 0.f) ? (int8_t)1 : (int8_t)-1;
}

__device__ __forceinline__ v8i_t load_afrag(const int8_t* rowp, int lo) {
  v8i_t a;
#pragma unroll
  for (int h = 0; h < 4; ++h) {
    uint2 x = *(const uint2*)(rowp + (h >> 1) * 32 + (h & 1) * 8 + 2 * lo);
    a[2 * h]     = (int)x.x;
    a[2 * h + 1] = (int)x.y;
  }
  return a;
}
__device__ __forceinline__ v8i_t load_bfrag(const int8_t* rowp, int koff) {
  uint4 x0 = *(const uint4*)(rowp + koff);
  uint4 x1 = *(const uint4*)(rowp + 32 + koff);
  v8i_t b;
  b[0] = (int)x0.x; b[1] = (int)x0.y; b[2] = (int)x0.z; b[3] = (int)x0.w;
  b[4] = (int)x1.x; b[5] = (int)x1.y; b[6] = (int)x1.z; b[7] = (int)x1.w;
  return b;
}

template <int CIN, int HIN, int WIN, int COUT, int HOUT, int WOUT>
__global__ __launch_bounds__(128)
void k_binconv_wmma(const int8_t* __restrict__ act, const int8_t* __restrict__ wsgn,
                    int* __restrict__ out) {
  constexpr int K   = CIN * 9;
  constexpr int NKT = K / 64;
  constexpr int PAD = 2;
  constexpr int HW  = HOUT * WOUT;
  __shared__ __align__(16) int8_t Alds[2][64][64];
  __shared__ __align__(16) int8_t Blds[2][64][64];

  const int tid  = threadIdx.x;
  const int wave = tid >> 5;
  const int lane = tid & 31;
  const int m0   = blockIdx.x * 64;
  const int n0   = blockIdx.y * 64;

  const int frow = tid >> 1;
  const int fseg = (tid & 1) * 32;
  const int M0  = m0 + frow;
  const int ab  = M0 / HW;
  const int rem = M0 - ab * HW;
  const int ay  = rem / WOUT;
  const int ax  = rem - ay * WOUT;

  unsigned av[8];
  uint4 bv0, bv1;

  auto stageA = [&](int kt) {
    int kb = kt * 64 + fseg;
#pragma unroll
    for (int d = 0; d < 8; ++d) {
      unsigned wpack = 0;
#pragma unroll
      for (int j = 0; j < 4; ++j) {
        int k  = kb + d * 4 + j;
        int ci = k / 9;
        int r  = k - ci * 9;
        int ky = r / 3;
        int kx = r - ky * 3;
        int iy = ay - PAD + ky;
        int ix = ax - PAD + kx;
        unsigned byteval = 0;
        if (iy >= 0 && iy < HIN && ix >= 0 && ix < WIN)
          byteval = (unsigned)(uint8_t)act[((size_t)(ab * CIN + ci) * HIN + iy) * WIN + ix];
        wpack |= byteval << (8 * j);
      }
      av[d] = wpack;
    }
  };
  auto stageB = [&](int kt) {
    const uint4* src = (const uint4*)(wsgn + (size_t)(n0 + frow) * K + kt * 64 + fseg);
    bv0 = src[0];
    bv1 = src[1];
    if (kt + 2 < NKT)
      __builtin_prefetch(wsgn + (size_t)(n0 + frow) * K + (kt + 2) * 64 + fseg, 0, 3);
  };
  auto commit = [&](int buf) {
    unsigned* ad = (unsigned*)&Alds[buf][frow][fseg];
#pragma unroll
    for (int d = 0; d < 8; ++d) ad[d] = av[d];
    uint4* bd = (uint4*)&Blds[buf][frow][fseg];
    bd[0] = bv0;
    bd[1] = bv1;
  };

  v8i_t acc[4] = {};
  stageA(0); stageB(0); commit(0);
  __syncthreads();

  const int lo   = (lane < 16) ? 0 : 8;
  const int koff = (lane < 16) ? 0 : 16;
  const int nl   = lane & 15;

  for (int kt = 0; kt < NKT; ++kt) {
    const int  cb   = kt & 1;
    const bool more = (kt + 1) < NKT;
    if (more) { stageA(kt + 1); stageB(kt + 1); }

    v8i_t bfrag = load_bfrag(&Blds[cb][wave * 16 + nl][0], koff);
#pragma unroll
    for (int ms = 0; ms < 4; ++ms) {
      v8i_t afrag = load_afrag(&Alds[cb][ms * 16 + nl][0], lo);
      acc[ms] = __builtin_amdgcn_wmma_i32_16x16x64_iu8(true, afrag, true, bfrag,
                                                       acc[ms], false, false);
    }
    if (more) commit((kt + 1) & 1);
    __syncthreads();
  }

  const int n = n0 + wave * 16 + nl;
#pragma unroll
  for (int ms = 0; ms < 4; ++ms) {
#pragma unroll
    for (int r = 0; r < 8; ++r) {
      int Mo = m0 + ms * 16 + r + ((lane < 16) ? 0 : 8);
      int b  = Mo / HW;
      int rm = Mo - b * HW;
      int y  = rm / WOUT;
      int x  = rm - y * WOUT;
      out[((size_t)(b * COUT + n) * HOUT + y) * WOUT + x] = acc[ms][r];
    }
  }
}

__global__ __launch_bounds__(128)
void k_binlinear_wmma(const int8_t* __restrict__ act, const int8_t* __restrict__ wsgn,
                      int* __restrict__ out, int M, int N, int K) {
  __shared__ __align__(16) int8_t Alds[2][64][64];
  __shared__ __align__(16) int8_t Blds[2][64][64];
  const int tid  = threadIdx.x;
  const int wave = tid >> 5;
  const int lane = tid & 31;
  const int m0   = blockIdx.x * 64;
  const int n0   = blockIdx.y * 64;
  const int frow = tid >> 1;
  const int fseg = (tid & 1) * 32;

  uint4 av0, av1, bv0, bv1;
  auto stage = [&](int kt) {
    const uint4* sa = (const uint4*)(act + (size_t)(m0 + frow) * K + kt * 64 + fseg);
    av0 = sa[0];
    av1 = sa[1];
    const uint4* sb = (const uint4*)(wsgn + (size_t)(n0 + frow) * K + kt * 64 + fseg);
    bv0 = sb[0];
    bv1 = sb[1];
  };
  auto commit = [&](int buf) {
    uint4* ad = (uint4*)&Alds[buf][frow][fseg];
    ad[0] = av0;
    ad[1] = av1;
    uint4* bd = (uint4*)&Blds[buf][frow][fseg];
    bd[0] = bv0;
    bd[1] = bv1;
  };

  v8i_t acc[4] = {};
  stage(0); commit(0);
  __syncthreads();

  const int lo   = (lane < 16) ? 0 : 8;
  const int koff = (lane < 16) ? 0 : 16;
  const int nl   = lane & 15;
  const int nkt  = K >> 6;

  for (int kt = 0; kt < nkt; ++kt) {
    const int  cb   = kt & 1;
    const bool more = (kt + 1) < nkt;
    if (more) stage(kt + 1);
    v8i_t bfrag = load_bfrag(&Blds[cb][wave * 16 + nl][0], koff);
#pragma unroll
    for (int ms = 0; ms < 4; ++ms) {
      v8i_t afrag = load_afrag(&Alds[cb][ms * 16 + nl][0], lo);
      acc[ms] = __builtin_amdgcn_wmma_i32_16x16x64_iu8(true, afrag, true, bfrag,
                                                       acc[ms], false, false);
    }
    if (more) commit((kt + 1) & 1);
    __syncthreads();
  }

  const int n = n0 + wave * 16 + nl;
#pragma unroll
  for (int ms = 0; ms < 4; ++ms) {
#pragma unroll
    for (int r = 0; r < 8; ++r) {
      int Mo = m0 + ms * 16 + r + ((lane < 16) ? 0 : 8);
      out[(size_t)Mo * N + n] = acc[ms][r];
    }
  }
}

__global__ void k_pool_bn_sign(const int* __restrict__ y, int8_t* __restrict__ a,
                               const float* __restrict__ g, const float* __restrict__ bb,
                               const float* __restrict__ mm, const float* __restrict__ vv,
                               int B, int C, int H, int W, int pool) {
  int Ho = H / pool, Wo = W / pool;
  int idx = blockIdx.x * blockDim.x + threadIdx.x;
  int total = B * C * Ho * Wo;
  if (idx >= total) return;
  int x = idx % Wo;
  int yy = (idx / Wo) % Ho;
  int c = (idx / (Wo * Ho)) % C;
  int b = idx / (Wo * Ho * C);
  const int* base = y + ((size_t)(b * C + c) * H + yy * pool) * W + x * pool;
  int vmax = base[0];
  for (int py = 0; py < pool; ++py)
    for (int px = 0; px < pool; ++px) {
      int t = base[py * W + px];
      vmax = t > vmax ? t : vmax;
    }
  float sc = g[c] * rsqrtf(vv[c] + EPSBN);
  float f = (float)vmax * sc + (bb[c] - mm[c] * sc);
  a[idx] = (f >= 0.f) ? (int8_t)1 : (int8_t)-1;
}

__global__ void k_bias_bn_sign(const int* __restrict__ z, const float* __restrict__ lb,
                               const float* __restrict__ g, const float* __restrict__ bb,
                               const float* __restrict__ mm, const float* __restrict__ vv,
                               int8_t* __restrict__ a, int M, int N) {
  int idx = blockIdx.x * blockDim.x + threadIdx.x;
  if (idx >= M * N) return;
  int n = idx % N;
  float sc = g[n] * rsqrtf(vv[n] + EPSBN);
  float h = (float)z[idx] + lb[n];
  float f = h * sc + (bb[n] - mm[n] * sc);
  a[idx] = (f >= 0.f) ? (int8_t)1 : (int8_t)-1;
}

__global__ void k_bias_bn_htanh(const int* __restrict__ z, const float* __restrict__ lb,
                                const float* __restrict__ g, const float* __restrict__ bb,
                                const float* __restrict__ mm, const float* __restrict__ vv,
                                float* __restrict__ o, int M, int N) {
  int idx = blockIdx.x * blockDim.x + threadIdx.x;
  if (idx >= M * N) return;
  int n = idx % N;
  float sc = g[n] * rsqrtf(vv[n] + EPSBN);
  float h = (float)z[idx] + lb[n];
  float f = h * sc + (bb[n] - mm[n] * sc);
  o[idx] = fminf(1.f, fmaxf(-1.f, f));
}

__global__ __launch_bounds__(64)
void k_final(const float* __restrict__ h, const float* __restrict__ w3,
             const float* __restrict__ b3,
             const float* __restrict__ g, const float* __restrict__ bb,
             const float* __restrict__ mm, const float* __restrict__ vv,
             float* __restrict__ out) {
  __shared__ float part[64][10];
  __shared__ float logits[10];
  int row = blockIdx.x;
  int tid = threadIdx.x;
  float acc[10];
#pragma unroll
  for (int j = 0; j < 10; ++j) acc[j] = 0.f;
  for (int k = tid; k < 1024; k += 64) {
    float a = h[(size_t)row * 1024 + k];
#pragma unroll
    for (int j = 0; j < 10; ++j) acc[j] += a * w3[j * 1024 + k];
  }
#pragma unroll
  for (int j = 0; j < 10; ++j) part[tid][j] = acc[j];
  __syncthreads();
  if (tid < 10) {
    float s = 0.f;
    for (int t = 0; t < 64; ++t) s += part[t][tid];
    s += b3[tid];
    float sc = g[tid] * rsqrtf(vv[tid] + EPSBN);
    logits[tid] = s * sc + (bb[tid] - mm[tid] * sc);
  }
  __syncthreads();
  if (tid < 10) {
    float mx = logits[0];
    for (int j = 1; j < 10; ++j) mx = fmaxf(mx, logits[j]);
    float se = 0.f;
    for (int j = 0; j < 10; ++j) se += expf(logits[j] - mx);
    out[row * 10 + tid] = logits[tid] - mx - logf(se);
  }
}

extern "C" void kernel_launch(void* const* d_in, const int* in_sizes, int n_in,
                              void* d_out, int out_size, void* d_ws, size_t ws_size,
                              hipStream_t stream) {
  (void)in_sizes; (void)n_in; (void)out_size; (void)ws_size;
  const float* x  = (const float*)d_in[0];
  const float* W0 = (const float*)d_in[1];
  const float* Wb[5] = {(const float*)d_in[2], (const float*)d_in[3],
                        (const float*)d_in[4], (const float*)d_in[5],
                        (const float*)d_in[6]};
  auto bng = [&](int i) { return (const float*)d_in[7 + 4 * i]; };
  auto bnb = [&](int i) { return (const float*)d_in[8 + 4 * i]; };
  auto bnm = [&](int i) { return (const float*)d_in[9 + 4 * i]; };
  auto bnv = [&](int i) { return (const float*)d_in[10 + 4 * i]; };
  const float* lw1 = (const float*)d_in[43];
  const float* lb1 = (const float*)d_in[44];
  const float* lw2 = (const float*)d_in[45];
  const float* lb2 = (const float*)d_in[46];
  const float* lw3 = (const float*)d_in[47];
  const float* lb3 = (const float*)d_in[48];

  char* ws = (char*)d_ws;
  size_t off = 0;
  auto carve = [&](size_t bytes) -> char* {
    char* p = ws + off;
    off += (bytes + 255) & ~(size_t)255;
    return p;
  };
  int8_t* w1s  = (int8_t*)carve(128 * 128 * 9);
  int8_t* w2s  = (int8_t*)carve(256 * 128 * 9);
  int8_t* w3s  = (int8_t*)carve(256 * 256 * 9);
  int8_t* w4s  = (int8_t*)carve(512 * 256 * 9);
  int8_t* w5s  = (int8_t*)carve(512 * 512 * 9);
  int8_t* lw1s = (int8_t*)carve((size_t)1024 * 25088);
  int8_t* lw2s = (int8_t*)carve((size_t)1024 * 1024);
  int8_t* actA = (int8_t*)carve((size_t)128 * 128 * 30 * 30);
  int8_t* actB = (int8_t*)carve((size_t)128 * 128 * 30 * 30);
  int*    ybuf = (int*)   carve((size_t)128 * 128 * 32 * 32 * 4);
  float*  h7   = (float*) carve((size_t)128 * 1024 * 4);

  auto signify = [&](const float* src, int8_t* dst, int n) {
    k_sign_i8<<<(n + 255) / 256, 256, 0, stream>>>(src, dst, n);
  };
  signify(Wb[0], w1s, 128 * 128 * 9);
  signify(Wb[1], w2s, 256 * 128 * 9);
  signify(Wb[2], w3s, 256 * 256 * 9);
  signify(Wb[3], w4s, 512 * 256 * 9);
  signify(Wb[4], w5s, 512 * 512 * 9);
  signify(lw1, lw1s, 1024 * 25088);
  signify(lw2, lw2s, 1024 * 1024);

  {
    int n = 128 * 128 * 30 * 30;
    k_conv0_bn_sign<<<(n + 255) / 256, 256, 0, stream>>>(
        x, W0, bng(0), bnb(0), bnm(0), bnv(0), actA);
  }

  auto poolbn = [&](int8_t* outact, int layer, int C, int H, int W, int pool) {
    int tot = 128 * C * (H / pool) * (W / pool);
    k_pool_bn_sign<<<(tot + 255) / 256, 256, 0, stream>>>(
        ybuf, outact, bng(layer), bnb(layer), bnm(layer), bnv(layer),
        128, C, H, W, pool);
  };

  k_binconv_wmma<128, 30, 30, 128, 32, 32>
      <<<dim3(131072 / 64, 128 / 64), 128, 0, stream>>>(actA, w1s, ybuf);
  poolbn(actB, 1, 128, 32, 32, 2);
  k_binconv_wmma<128, 16, 16, 256, 18, 18>
      <<<dim3(41472 / 64, 256 / 64), 128, 0, stream>>>(actB, w2s, ybuf);
  poolbn(actA, 2, 256, 18, 18, 1);
  k_binconv_wmma<256, 18, 18, 256, 20, 20>
      <<<dim3(51200 / 64, 256 / 64), 128, 0, stream>>>(actA, w3s, ybuf);
  poolbn(actB, 3, 256, 20, 20, 2);
  k_binconv_wmma<256, 10, 10, 512, 12, 12>
      <<<dim3(18432 / 64, 512 / 64), 128, 0, stream>>>(actB, w4s, ybuf);
  poolbn(actA, 4, 512, 12, 12, 1);
  k_binconv_wmma<512, 12, 12, 512, 14, 14>
      <<<dim3(25088 / 64, 512 / 64), 128, 0, stream>>>(actA, w5s, ybuf);
  poolbn(actB, 5, 512, 14, 14, 2);

  {
    dim3 g1(128 / 64, 1024 / 64);
    k_binlinear_wmma<<<g1, 128, 0, stream>>>(actB, lw1s, ybuf, 128, 1024, 25088);
    int tot = 128 * 1024;
    k_bias_bn_sign<<<(tot + 255) / 256, 256, 0, stream>>>(
        ybuf, lb1, bng(6), bnb(6), bnm(6), bnv(6), actA, 128, 1024);
    k_binlinear_wmma<<<g1, 128, 0, stream>>>(actA, lw2s, ybuf, 128, 1024, 1024);
    k_bias_bn_htanh<<<(tot + 255) / 256, 256, 0, stream>>>(
        ybuf, lb2, bng(7), bnb(7), bnm(7), bnv(7), h7, 128, 1024);
  }
  k_final<<<128, 64, 0, stream>>>(h7, lw3, lb3, bng(8), bnb(8), bnm(8), bnv(8),
                                  (float*)d_out);
}
